// RPN_47639777247769
// MI455X (gfx1250) — matrix operand order repaired and checked by reading, not compiled, not run on hardware
//
#include <hip/hip_runtime.h>
#include <hip/hip_bf16.h>
#include <cstdint>

typedef __attribute__((ext_vector_type(16))) _Float16 v16h;
typedef __attribute__((ext_vector_type(8)))  _Float16 v8h;
typedef __attribute__((ext_vector_type(8)))  float    v8f;

#define SUMHW3 159882
#define TOTSEL 4507
#define SORTN  8192
#define RPN_NMS_THR 0.7f
#define RPN_NEG_INF (-1000000000.0f)

__constant__ int c_HW3[5]  = {120000, 30000, 7500, 1875, 507};
__constant__ int c_DLVL[5] = {0, 120000, 150000, 157500, 159375};
__constant__ int c_K[5]    = {1000, 1000, 1000, 1000, 507};
__constant__ int c_SOFF[5] = {0, 1000, 2000, 3000, 4000};

// ---------------------------------------------------------------------------
// K0: repack conv weights fp32 OIHW -> f16 [oc][tap=ky*3+kx][ic]
// ---------------------------------------------------------------------------
__global__ void RPN_wcvt(const float* __restrict__ w, _Float16* __restrict__ W16)
{
    int i = blockIdx.x * blockDim.x + threadIdx.x;
    if (i >= 256 * 2304) return;
    int oc = i / 2304, r = i - oc * 2304;
    int tap = r >> 8, ic = r & 255;
    W16[i] = (_Float16)w[(size_t)oc * 2304 + ic * 9 + tap];
}

// ---------------------------------------------------------------------------
// K1: fused 3x3 conv (implicit GEMM on WMMA) + bias + ReLU + 1x1 heads
// block = 256 thr (8 waves). Tile: 16 pixels (one row) x 256 out channels.
// wave w handles oc [32w, 32w+32) via two 16x16x32 f16 WMMA accumulators.
// ---------------------------------------------------------------------------
__global__ __launch_bounds__(256)
void RPN_conv(const float* __restrict__ feat, const _Float16* __restrict__ W16,
              const float* __restrict__ cb,
              const float* __restrict__ lw, const float* __restrict__ lb,
              const float* __restrict__ dw, const float* __restrict__ db,
              float* __restrict__ sc, float* __restrict__ dl,
              int H, int W)
{
    const int tid = threadIdx.x;
    const int x0  = blockIdx.x * 16;
    const int y   = blockIdx.y;
    const int b   = blockIdx.z;

    // halo tile, NHWC f16, padded 256->264 halfs to spread LDS banks
    __shared__ __attribute__((aligned(32))) _Float16 xt[3][18][264];
    __shared__ __attribute__((aligned(32))) _Float16 hsh[16][264];

    for (int idx = tid; idx < 3 * 18 * 256; idx += 256) {
        int ch = idx / 54; int s = idx - ch * 54;
        int r  = s / 18;   int c = s - r * 18;
        int yy = y + r - 1, xx = x0 + c - 1;
        float v = 0.f;
        if (yy >= 0 && yy < H && xx >= 0 && xx < W)
            v = feat[(((size_t)b * 256 + ch) * H + yy) * W + xx];
        xt[r][c][ch] = (_Float16)v;
    }
    __syncthreads();

    const int lane = tid & 31;
    const int wv   = tid >> 5;
    const int h16  = lane >> 4;   // lane half (A/B K-striping)
    const int l16  = lane & 15;   // row (A) / column-pixel (B)
    const int ocb  = wv * 32;

    v8f acc0 = {}; v8f acc1 = {};

    for (int tap = 0; tap < 9; ++tap) {
        const int ky = tap / 3, kx = tap - ky * 3;
        const _Float16* bp  = &xt[ky][kx + l16][h16 * 8];
        const _Float16* a0p = W16 + (size_t)(ocb + l16) * 2304 + tap * 256 + h16 * 8;
        const _Float16* a1p = a0p + (size_t)16 * 2304;
        __builtin_prefetch(a0p + 256, 0, 0);   // global_prefetch_b8
#pragma unroll
        for (int ic0 = 0; ic0 < 256; ic0 += 32) {
            // B (32K x 16N): lane<16 holds K {0..7,16..23}; lane>=16 {8..15,24..31}
            v8h b0  = *(const v8h*)(bp + ic0);
            v8h b1  = *(const v8h*)(bp + ic0 + 16);
            // A (16M x 32K): lane<16 holds K {0..7,16..23}; lane>=16 {8..15,24..31}
            v8h al0 = *(const v8h*)(a0p + ic0);
            v8h ah0 = *(const v8h*)(a0p + ic0 + 16);
            v8h al1 = *(const v8h*)(a1p + ic0);
            v8h ah1 = *(const v8h*)(a1p + ic0 + 16);
            v16h A0, A1, Bm;
#pragma unroll
            for (int t = 0; t < 8; ++t) {
                A0[t] = al0[t]; A0[8 + t] = ah0[t];
                A1[t] = al1[t]; A1[8 + t] = ah1[t];
                Bm[t] = b0[t];  Bm[8 + t] = b1[t];
            }
            acc0 = __builtin_amdgcn_wmma_f32_16x16x32_f16(false, A0, false, Bm,
                                                          (short)0, acc0, false, false);
            acc1 = __builtin_amdgcn_wmma_f32_16x16x32_f16(false, A1, false, Bm,
                                                          (short)0, acc1, false, false);
        }
    }

    // D layout: VGPR r -> M = r + (lane>=16 ? 8 : 0), N = lane&15
#pragma unroll
    for (int r = 0; r < 8; ++r) {
        int m0 = ocb + r + h16 * 8;
        float v0 = acc0[r] + cb[m0]; v0 = v0 > 0.f ? v0 : 0.f;
        hsh[l16][m0] = (_Float16)v0;
        int m1 = m0 + 16;
        float v1 = acc1[r] + cb[m1]; v1 = v1 > 0.f ? v1 : 0.f;
        hsh[l16][m1] = (_Float16)v1;
    }
    __syncthreads();

    // 1x1 heads: 15 outputs (3 logits + 12 deltas) x 16 pixels
    if (tid < 240) {
        int p = tid & 15, o = tid >> 4;
        const float* wr; float bias;
        if (o < 3) { wr = lw + o * 256;       bias = lb[o]; }
        else       { wr = dw + (o - 3) * 256; bias = db[o - 3]; }
        float s = bias;
        for (int ch = 0; ch < 256; ++ch) s += wr[ch] * (float)hsh[p][ch];
        int x = x0 + p;
        if (x < W) {
            size_t sidx = (size_t)b * SUMHW3 + (size_t)(y * W + x) * 3;
            if (o < 3) sc[sidx + o] = s;
            else { int a = (o - 3) >> 2, j = (o - 3) & 3; dl[(sidx + a) * 4 + j] = s; }
        }
    }
}

// ---------------------------------------------------------------------------
// K2: deterministic radix-select top-K per (level,batch). One block each.
// ---------------------------------------------------------------------------
__device__ __forceinline__ unsigned RPN_fkey(float f)
{
    unsigned u = __float_as_uint(f);
    return (u & 0x80000000u) ? ~u : (u | 0x80000000u);  // monotone, bigger=bigger
}

__global__ __launch_bounds__(256)
void RPN_topk(const float* __restrict__ scores,
              int* __restrict__ sel_idx, float* __restrict__ sel_score)
{
    const int lvl = blockIdx.x, b = blockIdx.y;
    const int tid = threadIdx.x, bd = blockDim.x;
    const int N = c_HW3[lvl], K = c_K[lvl];
    const float* s = scores + (size_t)b * SUMHW3 + c_DLVL[lvl];

    __shared__ unsigned hist[256];
    __shared__ unsigned sh_prefix, sh_rem;
    __shared__ int sgt[257], seq[257];

    if (tid == 0) { sh_prefix = 0u; sh_rem = (unsigned)K; }
    __syncthreads();

    for (int shift = 24; shift >= 0; shift -= 8) {
        hist[tid] = 0u;
        __syncthreads();
        unsigned pfx = sh_prefix;
        unsigned hm  = (shift == 24) ? 0u : (0xFFFFFFFFu << (shift + 8));
        for (int i = tid; i < N; i += bd) {
            unsigned u = RPN_fkey(s[i]);
            if ((u & hm) == pfx) atomicAdd(&hist[(u >> shift) & 255], 1u);
        }
        __syncthreads();
        if (tid == 0) {
            unsigned rem = sh_rem, cum = 0; int bin = 255;
            for (; bin > 0; --bin) {
                if (cum + hist[bin] >= rem) break;
                cum += hist[bin];
            }
            sh_prefix = pfx | ((unsigned)bin << shift);
            sh_rem = rem - cum;
        }
        __syncthreads();
    }
    const unsigned T  = sh_prefix;       // exact key of K-th element
    const int     rem = (int)sh_rem;     // # of ==T elements to take
    const int     CG  = K - rem;         // # of  >T elements

    // deterministic compaction via per-thread rank (no atomic ordering)
    int mg = 0, me = 0;
    for (int i = tid; i < N; i += bd) {
        unsigned u = RPN_fkey(s[i]);
        mg += (u > T); me += (u == T);
    }
    sgt[tid + 1] = mg; seq[tid + 1] = me;
    __syncthreads();
    if (tid == 0) {
        sgt[0] = 0; seq[0] = 0;
        for (int t = 1; t <= bd; ++t) { sgt[t] += sgt[t - 1]; seq[t] += seq[t - 1]; }
    }
    __syncthreads();
    int pg = sgt[tid], pe = seq[tid];
    const size_t base = (size_t)b * TOTSEL + c_SOFF[lvl];
    for (int i = tid; i < N; i += bd) {
        unsigned u = RPN_fkey(s[i]);
        if (u > T) {
            sel_idx[base + pg] = i; sel_score[base + pg] = s[i]; ++pg;
        } else if (u == T) {
            if (pe < rem) { sel_idx[base + CG + pe] = i; sel_score[base + CG + pe] = s[i]; }
            ++pe;
        }
    }
}

// ---------------------------------------------------------------------------
// K3: decode + clip + validity mask
// ---------------------------------------------------------------------------
__global__ void RPN_decode(const int* __restrict__ sel_idx, const float* __restrict__ sel_score,
                           const float* a0, const float* a1, const float* a2,
                           const float* a3, const float* a4,
                           const float* __restrict__ dws,
                           float* __restrict__ boxes_cat, float* __restrict__ scores_cat)
{
    int g = blockIdx.x * blockDim.x + threadIdx.x;
    if (g >= 2 * TOTSEL) return;
    int b = g / TOTSEL, j = g - b * TOTSEL;
    int lvl = (j < 4000) ? (j / 1000) : 4;
    const float* anc = (lvl == 0) ? a0 : (lvl == 1) ? a1 : (lvl == 2) ? a2 : (lvl == 3) ? a3 : a4;
    int idx = sel_idx[g];
    const float* ap = anc + (size_t)idx * 4;
    const float* dp = dws + ((size_t)b * SUMHW3 + c_DLVL[lvl] + idx) * 4;
    float wa = ap[2] - ap[0], ha = ap[3] - ap[1];
    float xa = ap[0] + 0.5f * wa, ya = ap[1] + 0.5f * ha;
    float xc = dp[0] * wa + xa, yc = dp[1] * ha + ya;
    float w  = wa * expf(dp[2]), h = ha * expf(dp[3]);
    float b0 = xc - 0.5f * w, b1 = yc - 0.5f * h, b2 = xc + 0.5f * w, b3 = yc + 0.5f * h;
    b0 = fminf(fmaxf(b0, 0.f), 800.f); b1 = fminf(fmaxf(b1, 0.f), 800.f);
    b2 = fminf(fmaxf(b2, 0.f), 800.f); b3 = fminf(fmaxf(b3, 0.f), 800.f);
    bool valid = (b2 > b0) && (b3 > b1);
    boxes_cat[(size_t)g * 4 + 0] = b0; boxes_cat[(size_t)g * 4 + 1] = b1;
    boxes_cat[(size_t)g * 4 + 2] = b2; boxes_cat[(size_t)g * 4 + 3] = b3;
    scores_cat[g] = valid ? sel_score[g] : RPN_NEG_INF;
}

// ---------------------------------------------------------------------------
// K4/K5: stable descending bitonic sort (score, tiebreak idx asc) per batch
// ---------------------------------------------------------------------------
__global__ void RPN_sortinit(const float* __restrict__ scores_cat,
                             float* __restrict__ skey, int* __restrict__ sidx)
{
    int g = blockIdx.x * blockDim.x + threadIdx.x;
    if (g >= 2 * SORTN) return;
    int b = g >> 13, i = g & (SORTN - 1);
    skey[g] = (i < TOTSEL) ? scores_cat[(size_t)b * TOTSEL + i] : -3.0e38f;
    sidx[g] = i;
}

__global__ __launch_bounds__(1024)
void RPN_bitonic(float* __restrict__ skey, int* __restrict__ sidx)
{
    const int b = blockIdx.x;
    float* K_ = skey + (size_t)b * SORTN;
    int*   I_ = sidx + (size_t)b * SORTN;
    const unsigned tid = threadIdx.x, bd = blockDim.x;
    for (unsigned k = 2; k <= SORTN; k <<= 1) {
        for (unsigned j = k >> 1; j > 0; j >>= 1) {
            for (unsigned i = tid; i < SORTN; i += bd) {
                unsigned ixj = i ^ j;
                if (ixj > i) {
                    float ka = K_[i];   int ia = I_[i];
                    float kb = K_[ixj]; int ib = I_[ixj];
                    bool before = (ka > kb) || (ka == kb && ia < ib);
                    bool desc = ((i & k) == 0);
                    if (desc ? !before : before) {
                        K_[i] = kb; I_[i] = ib; K_[ixj] = ka; I_[ixj] = ia;
                    }
                }
            }
            __threadfence_block();
            __syncthreads();
        }
    }
}

__global__ void RPN_gatherb(const int* __restrict__ sidx, const float* __restrict__ boxes_cat,
                            float* __restrict__ bsort)
{
    int g = blockIdx.x * blockDim.x + threadIdx.x;
    if (g >= 2 * TOTSEL) return;
    int b = g / TOTSEL, i = g - b * TOTSEL;
    int si = sidx[(size_t)b * SORTN + i];
    const float* src = boxes_cat + ((size_t)b * TOTSEL + si) * 4;
    float* dst = bsort + (size_t)g * 4;
    dst[0] = src[0]; dst[1] = src[1]; dst[2] = src[2]; dst[3] = src[3];
}

// ---------------------------------------------------------------------------
// K6: greedy NMS + compact first 1000 kept boxes (zero-padded). 1 block/batch
// ---------------------------------------------------------------------------
__global__ __launch_bounds__(1024)
void RPN_nms_out(const float* __restrict__ bsort, float* __restrict__ out)
{
    const int b = blockIdx.x;
    const int tid = threadIdx.x, bd = blockDim.x;
    const float* B = bsort + (size_t)b * TOTSEL * 4;
    float* o = out + (size_t)b * 1000 * 4;
    __shared__ unsigned char keep[TOTSEL];
    __shared__ float bx[4];
    __shared__ int kf;
    for (int i = tid; i < TOTSEL; i += bd) keep[i] = 1;
    for (int i = tid; i < 4000;   i += bd) o[i] = 0.f;
    __syncthreads();
    for (int i = 0; i < TOTSEL; ++i) {
        if (tid == 0) {
            kf = keep[i];
            bx[0] = B[i * 4 + 0]; bx[1] = B[i * 4 + 1];
            bx[2] = B[i * 4 + 2]; bx[3] = B[i * 4 + 3];
        }
        __syncthreads();
        if (kf) {
            float ax0 = bx[0], ay0 = bx[1], ax1 = bx[2], ay1 = bx[3];
            float areaA = (ax1 - ax0) * (ay1 - ay0);
            for (int j = i + 1 + tid; j < TOTSEL; j += bd) {
                if (!keep[j]) continue;
                float jx0 = B[j * 4 + 0], jy0 = B[j * 4 + 1];
                float jx1 = B[j * 4 + 2], jy1 = B[j * 4 + 3];
                float iw = fminf(ax1, jx1) - fmaxf(ax0, jx0); iw = iw > 0.f ? iw : 0.f;
                float ih = fminf(ay1, jy1) - fmaxf(ay0, jy0); ih = ih > 0.f ? ih : 0.f;
                float inter = iw * ih;
                float areaJ = (jx1 - jx0) * (jy1 - jy0);
                float iou = inter / (areaA + areaJ - inter + 1e-9f);
                if (iou > RPN_NMS_THR) keep[j] = 0;
            }
        }
        __syncthreads();
    }
    if (tid == 0) {
        int cnt = 0;
        for (int i = 0; i < TOTSEL && cnt < 1000; ++i) {
            if (keep[i]) {
                o[cnt * 4 + 0] = B[i * 4 + 0]; o[cnt * 4 + 1] = B[i * 4 + 1];
                o[cnt * 4 + 2] = B[i * 4 + 2]; o[cnt * 4 + 3] = B[i * 4 + 3];
                ++cnt;
            }
        }
    }
}

// ---------------------------------------------------------------------------
extern "C" void kernel_launch(void* const* d_in, const int* in_sizes, int n_in,
                              void* d_out, int out_size, void* d_ws, size_t ws_size,
                              hipStream_t stream)
{
    (void)in_sizes; (void)n_in; (void)out_size; (void)ws_size;
    const float* feat[5] = {(const float*)d_in[0], (const float*)d_in[2], (const float*)d_in[4],
                            (const float*)d_in[6], (const float*)d_in[8]};
    const float* anc[5]  = {(const float*)d_in[1], (const float*)d_in[3], (const float*)d_in[5],
                            (const float*)d_in[7], (const float*)d_in[9]};
    const float* conv_w  = (const float*)d_in[10];
    const float* conv_b  = (const float*)d_in[11];
    const float* logit_w = (const float*)d_in[12];
    const float* logit_b = (const float*)d_in[13];
    const float* delta_w = (const float*)d_in[14];
    const float* delta_b = (const float*)d_in[15];
    float* out = (float*)d_out;

    char* ws = (char*)d_ws;
    size_t off = 0;
    auto take = [&](size_t bytes) -> char* {
        char* p = ws + off;
        off += (bytes + 255) & ~(size_t)255;
        return p;
    };
    _Float16* W16        = (_Float16*)take((size_t)256 * 2304 * 2);
    float*    scores_ws  = (float*)take((size_t)2 * SUMHW3 * 4);
    float*    deltas_ws  = (float*)take((size_t)2 * SUMHW3 * 16);
    int*      sel_idx    = (int*)take((size_t)2 * TOTSEL * 4);
    float*    sel_score  = (float*)take((size_t)2 * TOTSEL * 4);
    float*    boxes_cat  = (float*)take((size_t)2 * TOTSEL * 16);
    float*    scores_cat = (float*)take((size_t)2 * TOTSEL * 4);
    float*    skey       = (float*)take((size_t)2 * SORTN * 4);
    int*      sidx       = (int*)take((size_t)2 * SORTN * 4);
    float*    bsort      = (float*)take((size_t)2 * TOTSEL * 16);

    RPN_wcvt<<<(256 * 2304 + 255) / 256, 256, 0, stream>>>(conv_w, W16);

    static const int LH[5] = {200, 100, 50, 25, 13};
    static const int LW[5] = {200, 100, 50, 25, 13};
    static const int DLVL_h[5] = {0, 120000, 150000, 157500, 159375};
    for (int l = 0; l < 5; ++l) {
        dim3 grid((LW[l] + 15) / 16, LH[l], 2);
        RPN_conv<<<grid, 256, 0, stream>>>(feat[l], W16, conv_b,
                                           logit_w, logit_b, delta_w, delta_b,
                                           scores_ws + DLVL_h[l],
                                           deltas_ws + (size_t)DLVL_h[l] * 4,
                                           LH[l], LW[l]);
    }

    RPN_topk<<<dim3(5, 2), 256, 0, stream>>>(scores_ws, sel_idx, sel_score);
    RPN_decode<<<(2 * TOTSEL + 255) / 256, 256, 0, stream>>>(
        sel_idx, sel_score, anc[0], anc[1], anc[2], anc[3], anc[4],
        deltas_ws, boxes_cat, scores_cat);
    RPN_sortinit<<<(2 * SORTN + 255) / 256, 256, 0, stream>>>(scores_cat, skey, sidx);
    RPN_bitonic<<<2, 1024, 0, stream>>>(skey, sidx);
    RPN_gatherb<<<(2 * TOTSEL + 255) / 256, 256, 0, stream>>>(sidx, boxes_cat, bsort);
    RPN_nms_out<<<2, 1024, 0, stream>>>(bsort, out);
}
